// SimpleModel_47614007443740
// MI455X (gfx1250) — compile-verified, matrix operand order checked by reading, not run on hardware
//
#include <hip/hip_runtime.h>

typedef __attribute__((ext_vector_type(16))) _Float16 v16h;
typedef __attribute__((ext_vector_type(8)))  float    v8f;

static constexpr int kB   = 256;   // batch
static constexpr int kT   = 2048;  // sequence length (W of input)
static constexpr int kHgt = 32;    // height summed away
static constexpr int kH   = 11;    // LSTM hidden
static constexpr int kG   = 44;    // 4*hidden gate width

// ---------------- fast activations (v_exp_f32 / v_rcp_f32) ----------------
__device__ __forceinline__ float fexp(float x) {
    return __builtin_amdgcn_exp2f(x * 1.4426950408889634f);
}
__device__ __forceinline__ float fsig(float x) {
    return __builtin_amdgcn_rcpf(1.0f + fexp(-x));
}
__device__ __forceinline__ float ftanh(float x) {
    float cx = fminf(fmaxf(x, -9.0f), 9.0f);
    float e  = fexp(2.0f * cx);
    return (e - 1.0f) * __builtin_amdgcn_rcpf(e + 1.0f);
}

// ---------------- pass 1: height reduction, x(B,1,32,T) -> xs(B,T) --------
__global__ void __launch_bounds__(256) height_sum_kernel(
    const float* __restrict__ x, float* __restrict__ xs) {
    int idx = blockIdx.x * blockDim.x + threadIdx.x;   // b*T + t
    if (idx >= kB * kT) return;
    int b = idx / kT, t = idx % kT;
    const float* p = x + (size_t)b * kHgt * kT + t;
    float s = 0.f;
#pragma unroll
    for (int h = 0; h < kHgt; ++h) s += p[(size_t)h * kT];
    xs[idx] = s;
}

// A-matrix build: row r of padded hidden buffer (stride 32 floats, 16B aligned)
// lanes 0-15 carry K {0..7, 16..23}; lanes 16-31 carry K {8..15, 24..31}.
__device__ __forceinline__ v16h load_A(const float* __restrict__ rowbase, int hi8) {
    const float4* p = (const float4*)(rowbase + hi8);        // K = hi8 .. hi8+7
    const float4* q = (const float4*)(rowbase + 16 + hi8);   // K = 16+hi8 .. 16+hi8+7
    float4 f0 = p[0], f1 = p[1], f2 = q[0], f3 = q[1];
    v16h a;
    a[0]=(_Float16)f0.x; a[1]=(_Float16)f0.y; a[2]=(_Float16)f0.z; a[3]=(_Float16)f0.w;
    a[4]=(_Float16)f1.x; a[5]=(_Float16)f1.y; a[6]=(_Float16)f1.z; a[7]=(_Float16)f1.w;
    a[8]=(_Float16)f2.x; a[9]=(_Float16)f2.y; a[10]=(_Float16)f2.z; a[11]=(_Float16)f2.w;
    a[12]=(_Float16)f3.x; a[13]=(_Float16)f3.y; a[14]=(_Float16)f3.z; a[15]=(_Float16)f3.w;
    return a;
}

// ---------------- pass 2: fused 2-layer LSTM, one wave32 per 16-batch tile
__global__ void __launch_bounds__(32) lstm_fused_kernel(
    const float* __restrict__ xs,
    const float* __restrict__ W1, const float* __restrict__ U1, const float* __restrict__ b1,
    const float* __restrict__ W2, const float* __restrict__ U2, const float* __restrict__ b2,
    float* __restrict__ out) {

    // padded hidden buffers: row = batch-in-tile, 32 K-slots (extra slots stay 0)
    __shared__ __align__(16) float ha1[16 * 32];  // K: 0..10 = h1, 11 = x
    __shared__ __align__(16) float ha2[16 * 32];  // K: 0..10 = h2, 11..21 = h1_new

    const int lane = threadIdx.x;       // 0..31
    const int r    = lane & 15;         // tile row (M) for A / column (N) for B,C,D
    const int hi8  = (lane >> 4) << 3;  // 0 or 8
    const int b0   = blockIdx.x * 16;   // batch tile base

    for (int i = lane; i < 16 * 32; i += 32) { ha1[i] = 0.f; ha2[i] = 0.f; }
    __syncthreads();

    // ---- pack weights into WMMA B-matrix layout registers (held all 2048 steps)
    v16h Bw1[4], Bw2[4];
    float bias1[4], bias2[4];
    const bool nvalid = (r < kH);
#pragma unroll
    for (int g = 0; g < 4; ++g) {
        int j = g * kH + r;             // original gate column
        bias1[g] = nvalid ? b1[j] : 0.f;
        bias2[g] = nvalid ? b2[j] : 0.f;
#pragma unroll
        for (int e = 0; e < 16; ++e) {
            int s = e >> 1, hh = e & 1;
            int k = (s < 4) ? (s * 2 + hh + hi8) : (16 + (s - 4) * 2 + hh + hi8);
            float w1v = 0.f, w2v = 0.f;
            if (nvalid) {
                if (k < kH)            w1v = U1[k * kG + j];
                else if (k == kH)      w1v = W1[j];            // x row (D_in = 1)
                if (k < kH)            w2v = U2[k * kG + j];
                else if (k < 2 * kH)   w2v = W2[(k - kH) * kG + j];
            }
            Bw1[g][e] = (_Float16)w1v;
            Bw2[g][e] = (_Float16)w2v;
        }
    }

    // bias pre-splatted into C accumulators (D layout: 8 rows per lane)
    v8f cb1[4], cb2[4];
#pragma unroll
    for (int g = 0; g < 4; ++g)
#pragma unroll
        for (int v = 0; v < 8; ++v) { cb1[g][v] = bias1[g]; cb2[g][v] = bias2[g]; }

    v8f c1 = {}; v8f c2 = {};
    const float* xrow = xs + (size_t)(b0 + r) * kT;

#pragma unroll 1
    for (int t = 0; t < kT; ++t) {
        // stage x(t) into K-slot 11 of layer-1 A buffer
        float xv = xrow[t];
        __builtin_prefetch(xrow + t + 16, 0, 1);
        if (lane < 16) ha1[r * 32 + kH] = xv;
        __syncthreads();

        // ---------------- layer 1: gates = [h1, x] @ [U1; W1] + b1 ----------
        v16h a1 = load_A(&ha1[r * 32], hi8);
        v8f di = __builtin_amdgcn_wmma_f32_16x16x32_f16(false, a1, false, Bw1[0], (short)0, cb1[0], false, false);
        v8f df = __builtin_amdgcn_wmma_f32_16x16x32_f16(false, a1, false, Bw1[1], (short)0, cb1[1], false, false);
        v8f dg = __builtin_amdgcn_wmma_f32_16x16x32_f16(false, a1, false, Bw1[2], (short)0, cb1[2], false, false);
        v8f dq = __builtin_amdgcn_wmma_f32_16x16x32_f16(false, a1, false, Bw1[3], (short)0, cb1[3], false, false);

        v8f h1n;
#pragma unroll
        for (int v = 0; v < 8; ++v) {
            float iv = fsig(di[v]), fv = fsig(df[v]);
            float gv = ftanh(dg[v]), ov = fsig(dq[v]);
            float cc = fv * c1[v] + iv * gv;
            c1[v] = cc;
            h1n[v] = ov * ftanh(cc);
        }
        // publish h1: recurrent slot (ha1[0..10]) + layer-2 input slot (ha2[11..21])
        if (nvalid) {
#pragma unroll
            for (int v = 0; v < 8; ++v) {
                ha1[(v + hi8) * 32 + r] = h1n[v];
                ha2[(v + hi8) * 32 + kH + r] = h1n[v];
            }
        }
        __syncthreads();

        // ---------------- layer 2: gates = [h2, h1] @ [U2; W2] + b2 ---------
        v16h a2 = load_A(&ha2[r * 32], hi8);
        v8f ei = __builtin_amdgcn_wmma_f32_16x16x32_f16(false, a2, false, Bw2[0], (short)0, cb2[0], false, false);
        v8f ef = __builtin_amdgcn_wmma_f32_16x16x32_f16(false, a2, false, Bw2[1], (short)0, cb2[1], false, false);
        v8f eg = __builtin_amdgcn_wmma_f32_16x16x32_f16(false, a2, false, Bw2[2], (short)0, cb2[2], false, false);
        v8f eq = __builtin_amdgcn_wmma_f32_16x16x32_f16(false, a2, false, Bw2[3], (short)0, cb2[3], false, false);

        v8f h2n;
#pragma unroll
        for (int v = 0; v < 8; ++v) {
            float iv = fsig(ei[v]), fv = fsig(ef[v]);
            float gv = ftanh(eg[v]), ov = fsig(eq[v]);
            float cc = fv * c2[v] + iv * gv;
            c2[v] = cc;
            h2n[v] = ov * ftanh(cc);
        }
        // publish h2 recurrent state + write output (B, T, 11)
        if (nvalid) {
#pragma unroll
            for (int v = 0; v < 8; ++v) {
                ha2[(v + hi8) * 32 + r] = h2n[v];
                out[((size_t)(b0 + v + hi8) * kT + t) * kH + r] = h2n[v];
            }
        }
        __syncthreads();
    }
}

extern "C" void kernel_launch(void* const* d_in, const int* in_sizes, int n_in,
                              void* d_out, int out_size, void* d_ws, size_t ws_size,
                              hipStream_t stream) {
    (void)in_sizes; (void)n_in; (void)out_size; (void)ws_size;
    const float* x  = (const float*)d_in[0];
    const float* W1 = (const float*)d_in[1];
    const float* U1 = (const float*)d_in[2];
    const float* b1 = (const float*)d_in[3];
    const float* W2 = (const float*)d_in[4];
    const float* U2 = (const float*)d_in[5];
    const float* b2 = (const float*)d_in[6];
    float* xs  = (float*)d_ws;      // (B, T) height-summed sequence, 2 MB scratch
    float* out = (float*)d_out;     // (B, T, 11)

    height_sum_kernel<<<(kB * kT + 255) / 256, 256, 0, stream>>>(x, xs);
    lstm_fused_kernel<<<kB / 16, 32, 0, stream>>>(xs, W1, U1, b1, W2, U2, b2, out);
}
